// MKmeansNN_11665131176015
// MI455X (gfx1250) — matrix operand order repaired and checked by reading, not compiled, not run on hardware
//
#include <hip/hip_runtime.h>

typedef __attribute__((ext_vector_type(16))) __bf16 v16bf;
typedef __attribute__((ext_vector_type(8)))  __bf16 v8bf;
typedef __attribute__((ext_vector_type(8)))  float  v8f;

#define B_N 16384
#define M_N 16
#define K_N 256
#define D_N 64
#define ROW_DW 36   // DWORDs per LDS row: 32 data DWORDs (64 bf16) + 4 pad
                    // lane stride 36 (gcd 4 with 64 banks) -> conflict-free b128 reads

__device__ __forceinline__ unsigned int pack2(__bf16 a, __bf16 b) {
    unsigned short ua = __builtin_bit_cast(unsigned short, a);
    unsigned short ub = __builtin_bit_cast(unsigned short, b);
    return (unsigned int)ua | ((unsigned int)ub << 16);
}

__launch_bounds__(256)
__global__ void mkmeans_wmma_bf16_kernel(const float* __restrict__ x,
                                         const float* __restrict__ center,
                                         float* __restrict__ recon,
                                         float* __restrict__ center_out,
                                         float* __restrict__ label_out)
{
    __shared__ unsigned int clh[K_N * ROW_DW];  // bf16 "hi" plane of center[m]
    __shared__ unsigned int cll[K_N * ROW_DW];  // bf16 "lo" (residual) plane
    __shared__ float ncsq[K_N];                 // -||c_k||^2 (exact f32)

    const int m    = blockIdx.y;
    const int t    = threadIdx.x;
    const int lane = t & 31;
    const int wave = t >> 5;
    const int n    = lane & 15;   // column class (batch row for A, cluster col for B/D)
    const int h    = lane >> 4;   // wave half

    const float* cm = center + (size_t)m * K_N * D_N;

    // ---- issue x loads early; latency overlaps the LDS staging below ----
    // A 16-bit layout (16x32): lane half h, element j -> K = (j<8) ? h*8+j : 16+h*8+(j-8)
    const int b0 = (blockIdx.x * 8 + wave) * 16;
    const float* xr = x + ((size_t)(b0 + n) * M_N + m) * D_N;
    float4 xa[2][4];
    #pragma unroll
    for (int c = 0; c < 2; ++c) {
        xa[c][0] = *(const float4*)(xr + 32*c + 8*h);
        xa[c][1] = *(const float4*)(xr + 32*c + 8*h + 4);
        xa[c][2] = *(const float4*)(xr + 32*c + 16 + 8*h);
        xa[c][3] = *(const float4*)(xr + 32*c + 16 + 8*h + 4);
    }

    // ---- stage center[m] as bf16 hi/lo planes; exact f32 csq via butterfly ----
    // wave w handles rows w*32..w*32+31; lane l converts/stores DWORD l of the row
    for (int it = 0; it < 32; ++it) {
        const int k = wave * 32 + it;
        const float2 cv = *(const float2*)(cm + (size_t)k * D_N + 2 * lane);
        const __bf16 h0 = (__bf16)cv.x;
        const __bf16 h1 = (__bf16)cv.y;
        const __bf16 l0 = (__bf16)(cv.x - (float)h0);
        const __bf16 l1 = (__bf16)(cv.y - (float)h1);
        clh[k * ROW_DW + lane] = pack2(h0, h1);
        cll[k * ROW_DW + lane] = pack2(l0, l1);
        float s = cv.x * cv.x + cv.y * cv.y;
        #pragma unroll
        for (int mask = 1; mask <= 16; mask <<= 1) s += __shfl_xor(s, mask, 32);
        if (lane == 0) ncsq[k] = -s;
    }

    // ---- pass-through copy of center (one block column per m) ----
    if (blockIdx.x == 0) {
        float4* dst = (float4*)(center_out + (size_t)m * K_N * D_N);
        const float4* src = (const float4*)cm;
        for (int i = t; i < K_N * D_N / 4; i += 256) dst[i] = src[i];
    }

    // ---- split A into bf16 hi/lo with the 2x dot-scale folded in (exact) ----
    v16bf Ah[2], Al[2];
    #pragma unroll
    for (int c = 0; c < 2; ++c) {
        const float* f = (const float*)&xa[c][0];   // run order matches j mapping
        #pragma unroll
        for (int j = 0; j < 16; ++j) {
            const float v2 = 2.0f * f[j];
            const __bf16 hb = (__bf16)v2;
            Ah[c][j] = hb;
            Al[c][j] = (__bf16)(v2 - (float)hb);
        }
    }

    __syncthreads();

    float best[8];
    int   bidx[8];
    #pragma unroll
    for (int r = 0; r < 8; ++r) { best[r] = -3.0e38f; bidx[r] = 0; }

    // ---- score = 2*dot - csq; acc seeded with -csq; 6 bf16 WMMAs per k-tile ----
    for (int kt = 0; kt < 16; ++kt) {
        const int kb = kt * 16;
        const float nc = ncsq[kb + n];
        v8f acc;
        #pragma unroll
        for (int r = 0; r < 8; ++r) acc[r] = nc;

        // B 16-bit layout (32x16): lane (n,h), element j -> K = (j<8) ? h*8+j : 16+h*8+(j-8), col = n
        const __bf16* rowh = (const __bf16*)clh + (size_t)(kb + n) * (ROW_DW * 2) + h * 8;
        const __bf16* rowl = (const __bf16*)cll + (size_t)(kb + n) * (ROW_DW * 2) + h * 8;
        #pragma unroll
        for (int c = 0; c < 2; ++c) {
            const v8bf bh0 = *(const v8bf*)(rowh + c * 32);
            const v8bf bh1 = *(const v8bf*)(rowh + c * 32 + 16);
            const v8bf bl0 = *(const v8bf*)(rowl + c * 32);
            const v8bf bl1 = *(const v8bf*)(rowl + c * 32 + 16);
            const v16bf bh = __builtin_shufflevector(bh0, bh1,
                0,1,2,3,4,5,6,7,8,9,10,11,12,13,14,15);
            const v16bf bl = __builtin_shufflevector(bl0, bl1,
                0,1,2,3,4,5,6,7,8,9,10,11,12,13,14,15);
            acc = __builtin_amdgcn_wmma_f32_16x16x32_bf16(
                false, Ah[c], false, bh, (short)0, acc, false, false);
            acc = __builtin_amdgcn_wmma_f32_16x16x32_bf16(
                false, Ah[c], false, bl, (short)0, acc, false, false);
            acc = __builtin_amdgcn_wmma_f32_16x16x32_bf16(
                false, Al[c], false, bh, (short)0, acc, false, false);
        }

        #pragma unroll
        for (int r = 0; r < 8; ++r) {
            if (acc[r] > best[r]) { best[r] = acc[r]; bidx[r] = kb + n; }
        }
    }

    // ---- cross-lane argmax within each 16-lane half (rows r + 8h), tie -> lower k ----
    #pragma unroll
    for (int r = 0; r < 8; ++r) {
        float bv = best[r];
        int   bi = bidx[r];
        #pragma unroll
        for (int mask = 1; mask <= 8; mask <<= 1) {
            float ov = __shfl_xor(bv, mask, 32);
            int   oi = __shfl_xor(bi, mask, 32);
            if (ov > bv || (ov == bv && oi < bi)) { bv = ov; bi = oi; }
        }
        best[r] = bv; bidx[r] = bi;   // uniform across the 16 lanes of this half
    }

    // ---- labels (as float; one lane per row) ----
    #pragma unroll
    for (int r = 0; r < 8; ++r) {
        if (n == r) {
            label_out[(size_t)(b0 + r + 8 * h) * M_N + m] = (float)bidx[r];
        }
    }

    // ---- recon: gather winning center row (exact f32, L2-resident), coalesced ----
    #pragma unroll
    for (int r = 0; r < 8; ++r) {
        const int idx = bidx[r];
        const float4 v = *(const float4*)(cm + (size_t)idx * D_N + 4 * n);
        float4* dst = (float4*)(recon + ((size_t)(b0 + r + 8 * h) * M_N + m) * D_N);
        dst[n] = v;
    }
}

extern "C" void kernel_launch(void* const* d_in, const int* in_sizes, int n_in,
                              void* d_out, int out_size, void* d_ws, size_t ws_size,
                              hipStream_t stream) {
    (void)in_sizes; (void)n_in; (void)d_ws; (void)ws_size; (void)out_size;
    const float* x      = (const float*)d_in[0];
    const float* center = (const float*)d_in[1];

    float* recon      = (float*)d_out;                              // [B, M, D]
    float* center_out = recon + (size_t)B_N * M_N * D_N;            // [M, K, D]
    float* label_out  = center_out + (size_t)M_N * K_N * D_N;       // [B, M, 1]

    dim3 grid(B_N / (16 * 8), M_N);   // 128 x 16 blocks; 8 waves/block, 16 rows/wave
    mkmeans_wmma_bf16_kernel<<<grid, dim3(256), 0, stream>>>(
        x, center, recon, center_out, label_out);
}
